// AttentionModule_36842229465649
// MI455X (gfx1250) — matrix operand order repaired and checked by reading, not compiled, not run on hardware
//
#include <hip/hip_runtime.h>

// Single-head causal attention for MI455X (gfx1250, wave32, WMMA).
// B=4, S=4096, E=512, DK=DV=64. fp32 in/out, f16 WMMA with f32 accumulate.
//
// Data path: every WMMA operand fragment is exactly two ds_load_b128.
// A-operands read row-major tiles (CDNA5 A layout: k = hiA*8 + h in two
// 8-half chunks).  B-operands read [N][K]-major tiles (lane fixes N, h walks
// k = hiB*8 + h in two 8-half chunks, as A) -> V and W staged transposed.
// Attention K/V staging uses GLOBAL_LOAD_ASYNC_TO_LDS_B128 (ASYNCcnt) with
// double buffering so the next tile streams into LDS under current compute.

#define BATCH 4
#define SEQ   4096
#define EMB   512
#define HD    64

typedef _Float16 v16h __attribute__((ext_vector_type(16)));
typedef float    v8f  __attribute__((ext_vector_type(8)));
typedef _Float16 h8   __attribute__((ext_vector_type(8)));
typedef _Float16 h4   __attribute__((ext_vector_type(4)));
typedef float    f4   __attribute__((ext_vector_type(4)));

union frag16 { v16h v; h8 p[2]; };

__device__ __forceinline__ v8f wmma_f16(v16h a, v16h b, v8f c) {
  return __builtin_amdgcn_wmma_f32_16x16x32_f16(false, a, false, b, (short)0, c,
                                                false, false);
}

__device__ __forceinline__ v16h ld_a(const _Float16* row, int off) {
  frag16 f;
  f.p[0] = *(const h8*)(row + off);
  f.p[1] = *(const h8*)(row + off + 16);
  return f.v;
}

__device__ __forceinline__ v16h ld_b(const _Float16* nrow, int off) {
  frag16 f;
  f.p[0] = *(const h8*)(nrow + (off & ~31) + ((off >> 4) & 1) * 8);
  f.p[1] = *(const h8*)(nrow + (off & ~31) + ((off >> 4) & 1) * 8 + 16);
  return f.v;
}

// Async 16B global -> LDS copy (per lane). LDS byte offset = low 32 bits of
// the generic pointer (flat lowering places the LDS offset in addr[31:0]).
__device__ __forceinline__ void async_copy_b128(void* lds, const void* gptr) {
  const unsigned           l = (unsigned)(uintptr_t)lds;
  const unsigned long long g = (unsigned long long)(uintptr_t)gptr;
  asm volatile("global_load_async_to_lds_b128 %0, %1, off"
               :
               : "v"(l), "v"(g)
               : "memory");
}
__device__ __forceinline__ void wait_async_le2() {
  asm volatile("s_wait_asynccnt 0x2" ::: "memory");
}
__device__ __forceinline__ void wait_async_0() {
  asm volatile("s_wait_asynccnt 0x0" ::: "memory");
}

// ---------------------------------------------------------------------------
// Kernel 0a: W (E x HD, f32) -> Wt (HD x E, f16), for q/k/v.  Tiny (192 KB).
// ---------------------------------------------------------------------------
__global__ __launch_bounds__(256) void prep_wt_kernel(
    const float* __restrict__ Wq, const float* __restrict__ Wk,
    const float* __restrict__ Wv, _Float16* __restrict__ Wt) {
  const int which = blockIdx.y;
  const float* W = (which == 0) ? Wq : (which == 1) ? Wk : Wv;
  _Float16* Wo = Wt + (size_t)which * HD * EMB;
  const int g = blockIdx.x * 256 + threadIdx.x;  // over E*HD
  const int e = g >> 6, c = g & 63;
  Wo[(size_t)c * EMB + e] = (_Float16)W[g];
}

// ---------------------------------------------------------------------------
// Kernel 1: fused QKV projection.  grid=(B*S/32, 3), block=256 (8 waves).
// ---------------------------------------------------------------------------
__global__ __launch_bounds__(256) void qkv_proj_kernel(
    const float* __restrict__ X, const _Float16* __restrict__ Wt,
    const float* __restrict__ Bq, const float* __restrict__ Bk,
    const float* __restrict__ Bv,
    _Float16* __restrict__ Qh, _Float16* __restrict__ Kh,
    _Float16* __restrict__ Vh) {
  const int tid   = threadIdx.x;
  const int lane  = tid & 31;
  const int wv    = tid >> 5;
  const int m_sub = wv >> 2;  // 0..1
  const int n_sub = wv & 3;   // 0..3
  const int row0  = blockIdx.x * 32;
  const int which = blockIdx.y;

  const _Float16* Wm = Wt + (size_t)which * HD * EMB;  // [HD][EMB]
  const float*    Bm = (which == 0) ? Bq : (which == 1) ? Bk : Bv;
  _Float16*       Om = (which == 0) ? Qh : (which == 1) ? Kh : Vh;

  __shared__ _Float16 Xs[32][40];  // [m][k], pitch 80B
  __shared__ _Float16 Ws[64][40];  // [n][k], pitch 80B

  v8f c = {};

  const int xr = tid >> 3, xc = (tid & 7) * 4;  // X: 4 f32 -> 4 f16
  const int wr = tid >> 2, wc = (tid & 3) * 8;  // Wt: 8 f16

  const int rowA  = lane & 15;
  const int hiA8  = (lane >> 4) * 8;
  const int colB  = lane & 15;
  const int hiB16 = (lane >> 4) * 16;

  for (int kc = 0; kc < EMB / 32; ++kc) {
    const int k0 = kc * 32;
    {
      const f4 xv = *(const f4*)&X[(size_t)(row0 + xr) * EMB + k0 + xc];
      *(h4*)&Xs[xr][xc] = __builtin_convertvector(xv, h4);
      *(h8*)&Ws[wr][wc] = *(const h8*)&Wm[(size_t)wr * EMB + k0 + wc];
    }
    if (kc + 1 < EMB / 32)
      __builtin_prefetch(&X[(size_t)(row0 + xr) * EMB + k0 + 32 + xc], 0, 3);
    __syncthreads();

    const v16h a  = ld_a(&Xs[m_sub * 16 + rowA][0], hiA8);
    const v16h bf = ld_b(&Ws[n_sub * 16 + colB][0], hiB16);
    c = wmma_f16(a, bf, c);
    __syncthreads();
  }

  const int   col  = n_sub * 16 + (lane & 15);
  const float bias = Bm[col];
  const int   hi8  = (lane >> 4) << 3;
#pragma unroll
  for (int r = 0; r < 8; ++r) {
    const int m = m_sub * 16 + r + hi8;  // C/D: M = r (+8 for upper lanes)
    Om[(size_t)(row0 + m) * HD + col] = (_Float16)(c[r] + bias);
  }
}

// ---------------------------------------------------------------------------
// Kernel 1b: V [S][64] -> Vt [64][S] per batch (LDS tile transpose).
// ---------------------------------------------------------------------------
__global__ __launch_bounds__(256) void v_transpose_kernel(
    const _Float16* __restrict__ Vh, _Float16* __restrict__ Vt) {
  const int tid = threadIdx.x;
  const int s0  = blockIdx.x * 64;
  const int b   = blockIdx.y;
  const _Float16* Vb = Vh + (size_t)b * SEQ * HD;
  _Float16*       Tb = Vt + (size_t)b * HD * SEQ;

  __shared__ _Float16 Ts[64][72];

  {
    const int r = tid >> 2, c0 = (tid & 3) * 16;
    *(h8*)&Ts[r][c0]     = *(const h8*)&Vb[(size_t)(s0 + r) * HD + c0];
    *(h8*)&Ts[r][c0 + 8] = *(const h8*)&Vb[(size_t)(s0 + r) * HD + c0 + 8];
  }
  __syncthreads();
  {
    const int dv = tid >> 2, c0 = (tid & 3) * 16;
    h8 o0, o1;
#pragma unroll
    for (int j = 0; j < 8; ++j) o0[j] = Ts[c0 + j][dv];
#pragma unroll
    for (int j = 0; j < 8; ++j) o1[j] = Ts[c0 + 8 + j][dv];
    *(h8*)&Tb[(size_t)dv * SEQ + s0 + c0]     = o0;
    *(h8*)&Tb[(size_t)dv * SEQ + s0 + c0 + 8] = o1;
  }
}

// ---------------------------------------------------------------------------
// Kernel 2: causal flash attention.  grid=(S/64, B), block=256 (8 waves).
// 64-query tile; 32-key tiles, double-buffered via async global->LDS copies.
// ---------------------------------------------------------------------------
__global__ __launch_bounds__(256) void flash_attn_kernel(
    const _Float16* __restrict__ Qh, const _Float16* __restrict__ Kh,
    const _Float16* __restrict__ Vt, float* __restrict__ Out) {
  const int tid  = threadIdx.x;
  const int lane = tid & 31;
  const int wv   = tid >> 5;  // 0..7
  const int q0   = blockIdx.x * 64;
  const int bidx = blockIdx.y;

  const _Float16* Qb = Qh + (size_t)bidx * SEQ * HD;
  const _Float16* Kb = Kh + (size_t)bidx * SEQ * HD;
  const _Float16* Tb = Vt + (size_t)bidx * HD * SEQ;  // [dv][s]
  float*          Ob = Out + (size_t)bidx * SEQ * HD;

  __shared__ _Float16 Qs[64][72];       // [q][d] row-major (A operand)
  __shared__ _Float16 Ks[2][32][72];    // [key][d] = [N][K], double-buffered
  __shared__ _Float16 Vts[2][64][40];   // [dv][key] = [N][K], double-buffered
  __shared__ float    Ss[64][36];       // scores f32
  __shared__ _Float16 Ps[64][40];       // exp(scores) f16 (A of PV)
  __shared__ float row_max[64], row_sum[64], row_corr[64];

  const int qr = tid >> 2, qc = (tid & 3) * 16;    // Q staging coords
  const int kr = tid >> 3, kc = (tid & 7) * 8;     // K staging coords
  const int dv = tid >> 2, vc = (tid & 3) * 8;     // Vt staging coords

  // Prologue: async-stage Q tile (64x64) and key-tile 0 into buffer 0.
  async_copy_b128(&Qs[qr][qc], &Qb[(size_t)(q0 + qr) * HD + qc]);
  async_copy_b128(&Qs[qr][qc + 8], &Qb[(size_t)(q0 + qr) * HD + qc + 8]);
  async_copy_b128(&Ks[0][kr][kc], &Kb[(size_t)kr * HD + kc]);
  async_copy_b128(&Vts[0][dv][vc], &Tb[(size_t)dv * SEQ + vc]);
  if (tid < 64) { row_max[tid] = -1e30f; row_sum[tid] = 0.0f; }

  v8f acc0 = {}, acc1 = {};

  const int rowA  = lane & 15;
  const int hiA8  = (lane >> 4) * 8;
  const int colB  = lane & 15;
  const int hiB16 = (lane >> 4) * 16;
  const int sm    = wv >> 1, sn = wv & 1;  // score tile (4x2)
  const int m_sub = wv >> 1;               // PV row tile (0..3)
  const int n0    = (wv & 1) * 32;         // PV column base

  const int nkt = q0 / 32 + 2;  // causal: keys <= q0+63
  for (int kt = 0; kt < nkt; ++kt) {
    const int k0  = kt * 32;
    const int cur = kt & 1;

    // Issue async loads for the NEXT key tile into the spare buffer, then
    // wait until only those 2 ops remain outstanding (=> this tile resident).
    if (kt + 1 < nkt) {
      const int k1 = k0 + 32;
      async_copy_b128(&Ks[cur ^ 1][kr][kc], &Kb[(size_t)(k1 + kr) * HD + kc]);
      async_copy_b128(&Vts[cur ^ 1][dv][vc], &Tb[(size_t)dv * SEQ + k1 + vc]);
      wait_async_le2();
    } else {
      wait_async_0();
    }
    __syncthreads();

    // Scores: tile (sm,sn) = Q(16x64) . K^T(64x16), two chained WMMAs.
    v8f sc = {};
#pragma unroll
    for (int dc = 0; dc < 2; ++dc) {
      const int  d0 = dc * 32;
      const v16h a  = ld_a(&Qs[sm * 16 + rowA][0], d0 + hiA8);
      const v16h bf = ld_b(&Ks[cur][sn * 16 + colB][0], d0 + hiB16);
      sc = wmma_f16(a, bf, sc);
    }
    {
      const int scol = sn * 16 + (lane & 15);
      const int hi8  = (lane >> 4) << 3;
#pragma unroll
      for (int r = 0; r < 8; ++r)
        Ss[sm * 16 + r + hi8][scol] = sc[r] * 0.125f;  // 1/sqrt(64)
    }
    __syncthreads();

    // Online softmax: one thread per query row (64 rows).
    if (tid < 64) {
      const int qg   = q0 + tid;
      float     mold = row_max[tid];
      float     mnew = mold;
      float     s[32];
#pragma unroll
      for (int g = 0; g < 8; ++g) {
        const f4 v4 = *(const f4*)&Ss[tid][g * 4];
#pragma unroll
        for (int j = 0; j < 4; ++j) {
          const int   c2 = g * 4 + j;
          const float v  = ((k0 + c2) <= qg) ? v4[j] : -1e30f;
          s[c2] = v;
          mnew  = fmaxf(mnew, v);
        }
      }
      const float corr = __expf(mold - mnew);
      float       lsum = row_sum[tid] * corr;
#pragma unroll
      for (int g = 0; g < 4; ++g) {
        h8 pk;
#pragma unroll
        for (int j = 0; j < 8; ++j) {
          const float p = __expf(s[g * 8 + j] - mnew);
          pk[j] = (_Float16)p;
          lsum += p;
        }
        *(h8*)&Ps[tid][g * 8] = pk;
      }
      row_max[tid]  = mnew;
      row_sum[tid]  = lsum;
      row_corr[tid] = corr;
    }
    __syncthreads();

    // Rescale accumulators, then P(16x32) x V(32x32) per wave (2 WMMAs).
    {
      const int hi8 = (lane >> 4) << 3;
#pragma unroll
      for (int r = 0; r < 8; ++r) {
        const float cf = row_corr[m_sub * 16 + r + hi8];
        acc0[r] *= cf;
        acc1[r] *= cf;
      }
    }
    {
      const v16h a  = ld_a(&Ps[m_sub * 16 + rowA][0], hiA8);
      const v16h b0 = ld_b(&Vts[cur][n0 + colB][0], hiB16);
      const v16h b1 = ld_b(&Vts[cur][n0 + 16 + colB][0], hiB16);
      acc0 = wmma_f16(a, b0, acc0);
      acc1 = wmma_f16(a, b1, acc1);
    }
    __syncthreads();
  }

  // Normalize and store fp32 output.
  {
    const int hi8 = (lane >> 4) << 3;
#pragma unroll
    for (int r = 0; r < 8; ++r) {
      const int   mrow = m_sub * 16 + r + hi8;
      const float inv  = 1.0f / row_sum[mrow];
      Ob[(size_t)(q0 + mrow) * HD + n0 + colB]      = acc0[r] * inv;
      Ob[(size_t)(q0 + mrow) * HD + n0 + 16 + colB] = acc1[r] * inv;
    }
  }
}

// ---------------------------------------------------------------------------
extern "C" void kernel_launch(void* const* d_in, const int* in_sizes, int n_in,
                              void* d_out, int out_size, void* d_ws,
                              size_t ws_size, hipStream_t stream) {
  const float* X  = (const float*)d_in[0];
  const float* Wq = (const float*)d_in[1];
  const float* bq = (const float*)d_in[2];
  const float* Wk = (const float*)d_in[3];
  const float* bk = (const float*)d_in[4];
  const float* Wv = (const float*)d_in[5];
  const float* bv = (const float*)d_in[6];
  float*       out = (float*)d_out;

  const size_t NTOK = (size_t)BATCH * SEQ * HD;  // 1M elements
  _Float16* Qh  = (_Float16*)d_ws;  // 2 MB
  _Float16* Kh  = Qh + NTOK;        // 2 MB
  _Float16* Vh  = Kh + NTOK;        // 2 MB
  _Float16* Vtg = Vh + NTOK;        // 2 MB (V transposed per batch)
  _Float16* Wt  = Vtg + NTOK;       // 192 KB (3x Wt f16)

  dim3 g0(EMB * HD / 256, 3), b0(256);
  prep_wt_kernel<<<g0, b0, 0, stream>>>(Wq, Wk, Wv, Wt);

  dim3 g1(BATCH * SEQ / 32, 3), b1(256);
  qkv_proj_kernel<<<g1, b1, 0, stream>>>(X, Wt, bq, bk, bv, Qh, Kh, Vh);

  dim3 g1b(SEQ / 64, BATCH), b1b(256);
  v_transpose_kernel<<<g1b, b1b, 0, stream>>>(Vh, Vtg);

  dim3 g2(SEQ / 64, BATCH), b2(256);
  flash_attn_kernel<<<g2, b2, 0, stream>>>(Qh, Kh, Vtg, out);
}
